// Flow_Attention_10668698763928
// MI455X (gfx1250) — matrix operand order repaired and checked by reading, not compiled, not run on hardware
//
#include <hip/hip_runtime.h>
#include <hip/hip_bf16.h>

// ---------------------------------------------------------------------------
// Flow-Attention for MI455X (gfx1250, wave32, WMMA bf16, TDM warmup)
//   B=4, N=4096, C=1024, H=16, D=64
// ---------------------------------------------------------------------------

typedef __attribute__((ext_vector_type(16))) __bf16        v16bf;
typedef __attribute__((ext_vector_type(8)))  float         v8f;
typedef __attribute__((ext_vector_type(4)))  unsigned int  u32x4_t;
typedef __attribute__((ext_vector_type(4)))  int           i32x4_t;
typedef __attribute__((ext_vector_type(8)))  int           i32x8_t;

#define FA_B 4
#define FA_N 4096
#define FA_C 1024
#define FA_H 16
#define FA_D 64
#define FA_BH (FA_B * FA_H)          // 64
#define FA_M  (FA_B * FA_N)          // 16384 rows
#define FA_KVCH 16                   // N-chunks for the kv einsum
#define EPS 1e-6f

#if defined(__has_builtin)
#if __has_builtin(__builtin_amdgcn_tensor_load_to_lds)
#define FA_HAVE_TDM 1
#endif
#endif
#ifndef FA_HAVE_TDM
#define FA_HAVE_TDM 0
#endif

__device__ __forceinline__ float sigmoidf_(float x) {
  return 1.0f / (1.0f + __expf(-x));
}

__device__ __forceinline__ v8f wmma_bf16(v16bf a, v16bf b, v8f c) {
  return __builtin_amdgcn_wmma_f32_16x16x32_bf16(
      false, a, false, b, (short)0, c, false, false);
}

// pack two floats into one dword of bf16 pairs (lowers to v_cvt_pk_bf16_f32)
__device__ __forceinline__ unsigned pkbf(float lo, float hi) {
  union { __bf16 h[2]; unsigned u; } p;
  p.h[0] = (__bf16)lo;
  p.h[1] = (__bf16)hi;
  return p.u;
}

// WMMA A-fragment element mapping for K index kk (0..31):
//   lane_hi = (kk>>3)&1 ; elem = (kk&7) + (kk>=16 ? 8 : 0)
__device__ __forceinline__ int a_hi(int kk) { return (kk >> 3) & 1; }
__device__ __forceinline__ int a_e (int kk) { return (kk & 7) + ((kk & 16) ? 8 : 0); }
// Contiguous K mapping for kk (0..31), unused here (B frags use a_hi/a_e):
//   lane_hi = kk>=16 ; elem = kk&15
__device__ __forceinline__ int b_hi(int kk) { return (kk & 16) ? 1 : 0; }
__device__ __forceinline__ int b_e (int kk) { return kk & 15; }

// ---------------------------------------------------------------------------
// fp32 -> bf16 bulk convert (packed stores)
// ---------------------------------------------------------------------------
__global__ __launch_bounds__(256) void cvt_bf16_kernel(
    const float* __restrict__ src, __bf16* __restrict__ dst, int n4) {
  const int i = blockIdx.x * 256 + threadIdx.x;
  if (i < n4) {
    float4 f = ((const float4*)src)[i];
    ((uint2*)dst)[i] = make_uint2(pkbf(f.x, f.y), pkbf(f.z, f.w));
  }
}

// ---------------------------------------------------------------------------
// TDM warm-up: DMA a slab of the bf16 activations into LDS (pulls through
// L2), per CDNA5 ISA D# layout. Falls back to
// global_prefetch on toolchains without the builtin.
// ---------------------------------------------------------------------------
__global__ __launch_bounds__(32) void tdm_warm_kernel(
    const __bf16* __restrict__ src, float* __restrict__ dump) {
  __shared__ alignas(128) unsigned char stage[16384];
#if FA_HAVE_TDM
  unsigned long long ga =
      (unsigned long long)src + (unsigned long long)blockIdx.x * 16384ull;
  u32x4_t g0;
  g0[0] = 1u;                                     // count=1 user descriptor
  g0[1] = 0u;                                     // lds_addr = 0
  g0[2] = (unsigned)(ga & 0xffffffffu);           // global_addr[31:0]
  g0[3] = (unsigned)((ga >> 32) & 0x01ffffffu) |  // global_addr[56:32]
          (2u << 30);                              // type = 2 ("image")
  i32x8_t g1;
  const unsigned td0 = 4096;                      // tensor dim0 (elements)
  g1[0] = (1 << 16);                              // data_size = 2 bytes
  g1[1] = (int)((td0 & 0xffffu) << 16);           // tensor_dim0 lo16
  g1[2] = (int)(td0 >> 16) | (int)((1024u & 0xffffu) << 16); // dim0 hi / dim1 lo
  g1[3] = 0 | (int)(2048u << 16);                 // dim1 hi / tile_dim0 = 2048
  g1[4] = 4;                                      // tile_dim1 = 4 rows
  g1[5] = (int)td0;                               // tensor_dim0_stride lo32
  g1[6] = 0;
  g1[7] = 0;
  i32x4_t gz = {0, 0, 0, 0};
#if __clang_major__ >= 23
  i32x8_t gz8 = {0, 0, 0, 0, 0, 0, 0, 0};
  __builtin_amdgcn_tensor_load_to_lds(g0, g1, gz, gz, gz8, 0);
#else
  __builtin_amdgcn_tensor_load_to_lds(g0, g1, gz, gz, 0);
#endif
  __builtin_amdgcn_s_wait_tensorcnt(0);
  dump[blockIdx.x * 32 + threadIdx.x] = (float)stage[threadIdx.x];
#else
  __builtin_prefetch(src + (size_t)blockIdx.x * 8192 + threadIdx.x * 256, 0, 0);
  dump[blockIdx.x * 32 + threadIdx.x] = 0.0f;
  (void)stage;
#endif
}

// ---------------------------------------------------------------------------
// GEMM: Y = A(MxK bf16) @ W^T (W: NoutxK bf16) + bias, bf16 WMMA, f32 acc.
// MODE 0: plain f32 store (M x Nout)
// MODE 1: QKV epilogue -> sigmoid(q), sigmoid(k), v scattered to (B,H,N,D)
// Block tile 128x256, K-step 32, double-buffered fragment-order LDS tiles.
// 8 waves as 2(M)x4(N); each wave owns a 4x4 grid of 16x16 tiles:
// 16 WMMA per 16 ds_load_b128 per K-step.
// ---------------------------------------------------------------------------
template <int MODE>
__global__ __launch_bounds__(256) void wmma_gemm_kernel(
    const __bf16* __restrict__ A, const __bf16* __restrict__ W,
    const float* __restrict__ bias, float* __restrict__ out,
    float* __restrict__ qb, float* __restrict__ kb, float* __restrict__ vb,
    int M, int K, int Nout) {
  __shared__ alignas(32) __bf16 Als[2][8 * 32 * 16];   // 16KB
  __shared__ alignas(32) __bf16 Bls[2][16 * 32 * 16];  // 32KB

  const int t    = threadIdx.x;
  const int lane = t & 31;
  const int wave = t >> 5;
  const int bm   = blockIdx.x * 128;
  const int bn   = blockIdx.y * 256;

  // A staging: thread t stages fragment (tile=wave, lane)
  const int sr  = lane & 15;
  const int hiA = (lane & 16) ? 8 : 0;
  const int mA  = bm + wave * 16 + sr;

  const int wm = wave >> 2;  // 0..1
  const int wn = wave & 3;   // 0..3

  auto stageA = [&](int buf, int k0) {
    const __bf16* sa = A + (size_t)mA * K + k0 + hiA;
    uint4 x0 = *(const uint4*)sa;         // K offs +0..7
    uint4 x1 = *(const uint4*)(sa + 16);  // K offs +16..23
    uint4* dst = (uint4*)&Als[buf][(wave * 32 + lane) * 16];
    dst[0] = x0; dst[1] = x1;
  };
  auto stageB = [&](int buf, int k0) {
#pragma unroll
    for (int ss = 0; ss < 2; ++ss) {      // 512 fragment slots / 256 threads
      const int s  = t + ss * 256;
      const int jt = s >> 5, ln = s & 31;
      const int hiB = (ln & 16) ? 8 : 0;
      const int jB  = bn + jt * 16 + (ln & 15);
      const __bf16* sw = W + (size_t)jB * K + k0 + hiB;
      uint4 w0 = *(const uint4*)sw;       // K offs +0..7
      uint4 w1 = *(const uint4*)(sw + 16);  // K offs +16..23
      uint4* dst = (uint4*)&Bls[buf][(jt * 32 + ln) * 16];
      dst[0] = w0; dst[1] = w1;
    }
  };

  v8f acc[4][4] = {};

  stageA(0, 0);
  stageB(0, 0);
  __syncthreads();
  for (int k0 = 0; k0 < K; k0 += 32) {
    const int cur = (k0 >> 5) & 1;
    if (k0 + 32 < K) {
      stageA(cur ^ 1, k0 + 32);
      stageB(cur ^ 1, k0 + 32);
      __builtin_prefetch(A + (size_t)mA * K + k0 + 64, 0, 0);
    }
    v16bf af[4], bfr[4];
#pragma unroll
    for (int i = 0; i < 4; ++i)
      af[i] = *(const v16bf*)&Als[cur][((wm * 4 + i) * 32 + lane) * 16];
#pragma unroll
    for (int j = 0; j < 4; ++j)
      bfr[j] = *(const v16bf*)&Bls[cur][((wn * 4 + j) * 32 + lane) * 16];
#pragma unroll
    for (int i = 0; i < 4; ++i)
#pragma unroll
      for (int j = 0; j < 4; ++j)
        acc[i][j] = wmma_bf16(af[i], bfr[j], acc[i][j]);
    __syncthreads();
  }

  const int rlo = lane & 15;
  const int rhi = (lane & 16) ? 8 : 0;
#pragma unroll
  for (int i = 0; i < 4; ++i) {
    const int mbase = bm + (wm * 4 + i) * 16;
#pragma unroll
    for (int j = 0; j < 4; ++j) {
      const int jcol = bn + (wn * 4 + j) * 16 + rlo;
      const float bvl = bias[jcol];
#pragma unroll
      for (int vg = 0; vg < 8; ++vg) {
        const int m = mbase + vg + rhi;
        float val = acc[i][j][vg] + bvl;
        if (MODE == 0) {
          out[(size_t)m * Nout + jcol] = val;
        } else {
          const int which = jcol >> 10;  // 0=q 1=k 2=v
          const int c = jcol & 1023;
          const int h = c >> 6, d = c & 63;
          const int b = m >> 12, n = m & 4095;
          const size_t idx = ((((size_t)b * FA_H + h) * FA_N) + n) * FA_D + d;
          if (which == 0)       qb[idx] = sigmoidf_(val);
          else if (which == 1)  kb[idx] = sigmoidf_(val);
          else                  vb[idx] = val;
        }
      }
    }
  }
}

// ---------------------------------------------------------------------------
// Column sums over N per (b,h)
// ---------------------------------------------------------------------------
template <int USE_W>
__global__ __launch_bounds__(256) void colsum_kernel(
    const float* __restrict__ a, const float* __restrict__ b,
    const float* __restrict__ wa, const float* __restrict__ wb,
    float* __restrict__ outa, float* __restrict__ outb) {
  const int bh = blockIdx.x;
  const int t = threadIdx.x;
  const int d = t & 63, g = t >> 6;
  const float* ap = a + (size_t)bh * FA_N * FA_D;
  const float* bp = b + (size_t)bh * FA_N * FA_D;
  const float* wap = wa + (size_t)bh * FA_N;
  const float* wbp = wb + (size_t)bh * FA_N;
  float sa = 0.f, sb = 0.f;
  for (int n = g; n < FA_N; n += 4) {
    float fa = ap[(size_t)n * FA_D + d];
    float fb = bp[(size_t)n * FA_D + d];
    if (USE_W) { fa *= wap[n]; fb *= wbp[n]; }
    sa += fa; sb += fb;
  }
  __shared__ float la[256], lb[256];
  la[t] = sa; lb[t] = sb;
  __syncthreads();
  if (t < 64) {
    outa[bh * FA_D + t] = la[t] + la[t + 64] + la[t + 128] + la[t + 192];
    outb[bh * FA_D + t] = lb[t] + lb[t + 64] + lb[t + 128] + lb[t + 192];
  }
}

// ---------------------------------------------------------------------------
// per-row dots (wave per row, shfl_xor reduction)
// ---------------------------------------------------------------------------
__global__ __launch_bounds__(256) void pern_dot_kernel(
    const float* __restrict__ qb, const float* __restrict__ kb,
    const float* __restrict__ ksum, const float* __restrict__ qsum,
    float* __restrict__ si, float* __restrict__ so) {
  const int bh = blockIdx.x;
  const int wave = threadIdx.x >> 5, lane = threadIdx.x & 31;
  const int n = blockIdx.y * 8 + wave;
  const float* q = qb + ((size_t)bh * FA_N + n) * FA_D;
  const float* k = kb + ((size_t)bh * FA_N + n) * FA_D;
  const float* ks = ksum + bh * FA_D;
  const float* qs = qsum + bh * FA_D;
  const int d0 = lane * 2;
  float a = (q[d0] + EPS) * (ks[d0] + EPS) + (q[d0 + 1] + EPS) * (ks[d0 + 1] + EPS);
  float c = (k[d0] + EPS) * (qs[d0] + EPS) + (k[d0 + 1] + EPS) * (qs[d0 + 1] + EPS);
#pragma unroll
  for (int m = 16; m >= 1; m >>= 1) {
    a += __shfl_xor(a, m, 32);
    c += __shfl_xor(c, m, 32);
  }
  if (lane == 0) {
    si[(size_t)bh * FA_N + n] = 1.0f / a;
    so[(size_t)bh * FA_N + n] = 1.0f / c;
  }
}

__global__ __launch_bounds__(256) void conserved_kernel(
    const float* __restrict__ qb, const float* __restrict__ kb,
    const float* __restrict__ ksos, const float* __restrict__ qsis,
    float* __restrict__ salloc, float* __restrict__ csrc) {
  const int bh = blockIdx.x;
  const int wave = threadIdx.x >> 5, lane = threadIdx.x & 31;
  const int n = blockIdx.y * 8 + wave;
  const float* q = qb + ((size_t)bh * FA_N + n) * FA_D;
  const float* k = kb + ((size_t)bh * FA_N + n) * FA_D;
  const float* A = ksos + bh * FA_D;
  const float* Bv = qsis + bh * FA_D;
  const int d0 = lane * 2;
  float a = (q[d0] + EPS) * (A[d0] + EPS) + (q[d0 + 1] + EPS) * (A[d0 + 1] + EPS);
  float c = (k[d0] + EPS) * (Bv[d0] + EPS) + (k[d0 + 1] + EPS) * (Bv[d0 + 1] + EPS);
#pragma unroll
  for (int m = 16; m >= 1; m >>= 1) {
    a += __shfl_xor(a, m, 32);
    c += __shfl_xor(c, m, 32);
  }
  if (lane == 0) {
    salloc[(size_t)bh * FA_N + n] = sigmoidf_(a + EPS);  // ratio == 1
    float cc = c + EPS;
    cc = fminf(1.0f, fmaxf(-1.0f, cc));
    csrc[(size_t)bh * FA_N + n] = cc;
  }
}

__global__ __launch_bounds__(256) void softmax_kernel(
    const float* __restrict__ cs, float* __restrict__ sc) {
  const int bh = blockIdx.x;
  const int t = threadIdx.x;
  const float* x = cs + (size_t)bh * FA_N;
  __shared__ float red[256];
  float mx = -1e30f;
  for (int n = t; n < FA_N; n += 256) mx = fmaxf(mx, x[n]);
  red[t] = mx;
  __syncthreads();
  for (int s = 128; s > 0; s >>= 1) {
    if (t < s) red[t] = fmaxf(red[t], red[t + s]);
    __syncthreads();
  }
  mx = red[0];
  __syncthreads();
  float sm = 0.f;
  for (int n = t; n < FA_N; n += 256) sm += __expf(x[n] - mx);
  red[t] = sm;
  __syncthreads();
  for (int s = 128; s > 0; s >>= 1) {
    if (t < s) red[t] += red[t + s];
    __syncthreads();
  }
  sm = red[0];
  const float inv = (float)FA_N / sm;
  for (int n = t; n < FA_N; n += 256)
    sc[(size_t)bh * FA_N + n] = __expf(x[n] - mx) * inv;
}

// ---------------------------------------------------------------------------
// kv partials: kvpart[ch,bh,d,m] = sum_{n in chunk} k[n,d]*(v[n,m]*scomp[n])
// Fragment-order LDS tiles (scatter on store, contiguous v16bf on load).
// ---------------------------------------------------------------------------
__global__ __launch_bounds__(256) void kv_kernel(
    const float* __restrict__ kbuf, const float* __restrict__ vbuf,
    const float* __restrict__ scomp, float* __restrict__ kvpart) {
  __shared__ alignas(32) __bf16 Kf[4 * 32 * 16];  // A frags: [dt][lane][e]
  __shared__ alignas(32) __bf16 Vf[4 * 32 * 16];  // B frags: [mt][lane][e]
  const int bh = blockIdx.x;
  const int ch = blockIdx.y;
  const int t = threadIdx.x, lane = t & 31, wave = t >> 5;
  const float* kb = kbuf + (size_t)bh * FA_N * FA_D;
  const float* vb = vbuf + (size_t)bh * FA_N * FA_D;
  const float* sc = scomp + (size_t)bh * FA_N;
  const int nr = t >> 3;        // K index within 32-step
  const int d0 = (t & 7) * 8;   // 8 elements along D
  const int dt = wave & 3;
  const int mtb = (wave >> 2) * 2;
  const int ahi = a_hi(nr), ae = a_e(nr);
  const int bhi = a_hi(nr), be = a_e(nr);
  v8f acc[2] = {};
  const int nbase = ch * (FA_N / FA_KVCH);
  for (int it = 0; it < (FA_N / FA_KVCH) / 32; ++it) {
    const int n0 = nbase + it * 32;
    __syncthreads();
    {
      const float w = sc[n0 + nr];
      const float4* kp = (const float4*)(kb + (size_t)(n0 + nr) * FA_D + d0);
      const float4* vp = (const float4*)(vb + (size_t)(n0 + nr) * FA_D + d0);
      float4 k0v = kp[0], k1v = kp[1];
      float4 v0v = vp[0], v1v = vp[1];
      float kk[8] = {k0v.x, k0v.y, k0v.z, k0v.w, k1v.x, k1v.y, k1v.z, k1v.w};
      float vv[8] = {v0v.x, v0v.y, v0v.z, v0v.w, v1v.x, v1v.y, v1v.z, v1v.w};
#pragma unroll
      for (int j = 0; j < 8; ++j) {
        const int d = d0 + j;
        Kf[(((d >> 4) * 32) + ahi * 16 + (d & 15)) * 16 + ae] = (__bf16)kk[j];
        Vf[(((d >> 4) * 32) + bhi * 16 + (d & 15)) * 16 + be] = (__bf16)(vv[j] * w);
      }
    }
    __syncthreads();
    v16bf a  = *(const v16bf*)&Kf[(dt * 32 + lane) * 16];
    v16bf b0 = *(const v16bf*)&Vf[((mtb + 0) * 32 + lane) * 16];
    v16bf b1 = *(const v16bf*)&Vf[((mtb + 1) * 32 + lane) * 16];
    acc[0] = wmma_bf16(a, b0, acc[0]);
    acc[1] = wmma_bf16(a, b1, acc[1]);
  }
  const int rlo = lane & 15, rhi = (lane & 16) ? 8 : 0;
  float* dst = kvpart + (size_t)ch * (FA_BH * FA_D * FA_D) +
               (size_t)bh * FA_D * FA_D;
#pragma unroll
  for (int j = 0; j < 2; ++j)
#pragma unroll
    for (int vg = 0; vg < 8; ++vg) {
      const int d = dt * 16 + vg + rhi;
      const int m = (mtb + j) * 16 + rlo;
      dst[d * FA_D + m] = acc[j][vg];
    }
}

__global__ __launch_bounds__(256) void kv_reduce_kernel(
    const float* __restrict__ part, float* __restrict__ kv) {
  const int i = blockIdx.x * 256 + threadIdx.x;  // < 64*64*64
  float s = 0.f;
#pragma unroll
  for (int c = 0; c < FA_KVCH; ++c)
    s += part[(size_t)c * (FA_BH * FA_D * FA_D) + i];
  kv[i] = s;
}

// ---------------------------------------------------------------------------
// attn(bf16)[b,n,h*64+m] = salloc[n] * sum_d (q[n,d]*si[n]) * kv[d,m]
// kv pre-swizzled into fragment order in LDS once per block.
// ---------------------------------------------------------------------------
__global__ __launch_bounds__(256) void attnout_kernel(
    const float* __restrict__ qbuf, const float* __restrict__ kv,
    const float* __restrict__ si, const float* __restrict__ salloc,
    __bf16* __restrict__ attnh) {
  const int bh = blockIdx.x;
  const int b = bh >> 4, h = bh & 15;
  const int n0 = blockIdx.y * 128;
  __shared__ alignas(32) __bf16 Als[8 * 32 * 16];
  __shared__ alignas(32) __bf16 KVf[2 * 4 * 32 * 16];  // [k0blk][jt][lane][e]
  const int t = threadIdx.x, lane = t & 31, wave = t >> 5;
  const float* qb = qbuf + (size_t)bh * FA_N * FA_D;
  const float* sip = si + (size_t)bh * FA_N;
  const float* sap = salloc + (size_t)bh * FA_N;
  for (int i = t; i < FA_D * FA_D; i += 256) {
    const int d = i >> 6, m = i & 63;
    const int blk = d >> 5, kin = d & 31;
    const int hi = a_hi(kin), e = a_e(kin);
    const int jt = m >> 4, ll = m & 15;
    KVf[(((blk * 4 + jt) * 32) + hi * 16 + ll) * 16 + e] =
        (__bf16)kv[(size_t)bh * FA_D * FA_D + i];
  }
  const int sr = lane & 15;
  const int hiA = (lane & 16) ? 8 : 0;
  const int mrow = n0 + wave * 16 + sr;
  const float s = sip[mrow];
  v8f acc[4] = {};
#pragma unroll
  for (int k0 = 0; k0 < FA_D; k0 += 32) {
    __syncthreads();
    {
      const float* src = qb + (size_t)mrow * FA_D + k0 + hiA;
      float4 a0 = ((const float4*)src)[0];
      float4 a1 = ((const float4*)src)[1];
      float4 a2 = ((const float4*)(src + 16))[0];
      float4 a3 = ((const float4*)(src + 16))[1];
      uint4* dst = (uint4*)&Als[(wave * 32 + lane) * 16];
      dst[0] = make_uint4(pkbf(a0.x * s, a0.y * s), pkbf(a0.z * s, a0.w * s),
                          pkbf(a1.x * s, a1.y * s), pkbf(a1.z * s, a1.w * s));
      dst[1] = make_uint4(pkbf(a2.x * s, a2.y * s), pkbf(a2.z * s, a2.w * s),
                          pkbf(a3.x * s, a3.y * s), pkbf(a3.z * s, a3.w * s));
    }
    __syncthreads();
    v16bf a = *(const v16bf*)&Als[(wave * 32 + lane) * 16];
#pragma unroll
    for (int j = 0; j < 4; ++j) {
      v16bf bfr = *(const v16bf*)&KVf[(((k0 >> 5) * 4 + j) * 32 + lane) * 16];
      acc[j] = wmma_bf16(a, bfr, acc[j]);
    }
  }
  const int rlo = lane & 15, rhi = (lane & 16) ? 8 : 0;
#pragma unroll
  for (int j = 0; j < 4; ++j)
#pragma unroll
    for (int vg = 0; vg < 8; ++vg) {
      const int n = n0 + wave * 16 + vg + rhi;
      const int m = j * 16 + rlo;
      const float val = acc[j][vg] * sap[n];
      attnh[(((size_t)b * FA_N) + n) * FA_C + h * FA_D + m] = (__bf16)val;
    }
}

// ---------------------------------------------------------------------------
extern "C" void kernel_launch(void* const* d_in, const int* in_sizes, int n_in,
                              void* d_out, int out_size, void* d_ws,
                              size_t ws_size, hipStream_t stream) {
  const float* x      = (const float*)d_in[0];
  const float* qkv_w  = (const float*)d_in[1];
  const float* qkv_b  = (const float*)d_in[2];
  const float* proj_w = (const float*)d_in[3];
  const float* proj_b = (const float*)d_in[4];
  float* out = (float*)d_out;

  const size_t QSZ = (size_t)FA_BH * FA_N * FA_D;  // 16,777,216
  const size_t NBH = (size_t)FA_BH * FA_N;         // 262,144
  float* ws     = (float*)d_ws;
  float* q      = ws;
  float* k      = q + QSZ;
  float* v      = k + QSZ;
  float* qsum   = v + QSZ;
  float* ksum   = qsum + FA_BH * FA_D;
  float* qsis   = ksum + FA_BH * FA_D;
  float* ksos   = qsis + FA_BH * FA_D;
  float* si     = ksos + FA_BH * FA_D;
  float* so     = si + NBH;
  float* salloc = so + NBH;
  float* csrc   = salloc + NBH;
  float* scomp  = csrc + NBH;
  float* kvpart = scomp + NBH;                       // 16*64*64*64
  float* kvbuf  = kvpart + (size_t)FA_KVCH * FA_BH * FA_D * FA_D;
  float* fbase  = kvbuf + FA_BH * FA_D * FA_D;
  __bf16* xh     = (__bf16*)fbase;                   // 16M bf16
  __bf16* wqkvh  = xh + QSZ;                         // 3M bf16
  __bf16* wprojh = wqkvh + (size_t)3 * FA_C * FA_C;  // 1M bf16
  __bf16* attnh  = wprojh + (size_t)FA_C * FA_C;     // 16M bf16

  // 0) bf16 copies of GEMM operands
  cvt_bf16_kernel<<<(int)(QSZ / 4 / 256), 256, 0, stream>>>(x, xh, (int)(QSZ / 4));
  cvt_bf16_kernel<<<(3 * FA_C * FA_C / 4) / 256, 256, 0, stream>>>(
      qkv_w, wqkvh, 3 * FA_C * FA_C / 4);
  cvt_bf16_kernel<<<(FA_C * FA_C / 4) / 256, 256, 0, stream>>>(
      proj_w, wprojh, FA_C * FA_C / 4);

  // 0b) TDM warm of activations (guarded; also exercises s_wait_tensorcnt)
  tdm_warm_kernel<<<64, 32, 0, stream>>>(xh, kvpart);

  // 1) QKV projection + bias + sigmoid + (B,H,N,D) scatter
  wmma_gemm_kernel<1><<<dim3(FA_M / 128, (3 * FA_C) / 256), 256, 0, stream>>>(
      xh, wqkvh, qkv_b, nullptr, q, k, v, FA_M, FA_C, 3 * FA_C);

  // 2) q_sum, k_sum
  colsum_kernel<0><<<FA_BH, 256, 0, stream>>>(q, k, si, so, qsum, ksum);
  // 3) sink_incoming, source_outgoing
  pern_dot_kernel<<<dim3(FA_BH, FA_N / 8), 256, 0, stream>>>(q, k, ksum, qsum, si, so);
  // 4) k_so_sum, q_si_sum
  colsum_kernel<1><<<FA_BH, 256, 0, stream>>>(k, q, so, si, ksos, qsis);
  // 5) sink_allocation, clipped conserved_source
  conserved_kernel<<<dim3(FA_BH, FA_N / 8), 256, 0, stream>>>(q, k, ksos, qsis, salloc, csrc);
  // 6) softmax -> source_competition
  softmax_kernel<<<FA_BH, 256, 0, stream>>>(csrc, scomp);

  // 7) kv partials + reduce
  kv_kernel<<<dim3(FA_BH, FA_KVCH), 256, 0, stream>>>(k, v, scomp, kvpart);
  kv_reduce_kernel<<<(FA_BH * FA_D * FA_D) / 256, 256, 0, stream>>>(kvpart, kvbuf);

  // 8) attn = (q*si) @ kv * salloc -> bf16 (B,N,C)
  attnout_kernel<<<dim3(FA_BH, FA_N / 128), 256, 0, stream>>>(q, kvbuf, si, salloc, attnh);

  // 9) final projection
  wmma_gemm_kernel<0><<<dim3(FA_M / 128, FA_C / 256), 256, 0, stream>>>(
      attnh, wprojh, proj_b, out, nullptr, nullptr, nullptr, FA_M, FA_C, FA_C);
}
